// GNNNet_24137716203646
// MI455X (gfx1250) — matrix operand order repaired and checked by reading, not compiled, not run on hardware
//
#include <hip/hip_runtime.h>
#include <math.h>

typedef __attribute__((ext_vector_type(16))) _Float16 v16h;
typedef __attribute__((ext_vector_type(8)))  _Float16 v8h;
typedef __attribute__((ext_vector_type(4)))  _Float16 v4h;
typedef __attribute__((ext_vector_type(8)))  float    v8f;

#define HEADS 3
#define N_MOL 60000
#define E_MOL 480000
#define N_PRO 120000
#define E_PRO 720000
#define NB    256
#define FM    78
#define FP    54
#define OD    128

#define BM 128
#define BN 64
#define BK 32
#define LDS_STRIDE 40   // halves per row: 32 data + 8 pad (80B) -> conflict-free frag loads
#define CHUNK 24576

// A-fragment K permutation: lanes 0-15 need K {0..7,16..23}, lanes 16-31 need
// K {8..15,24..31}; storing k with bits 3<->4 swapped makes each lane's 16
// halves contiguous in LDS.
__device__ __forceinline__ int permk(int k) {
    return (k & ~24) | ((k & 8) << 1) | ((k & 16) >> 1);
}

// ---------------------------------------------------------------------------
// WMMA f16 GEMM: C[M,N] = A[M,K] * B[K,N] (+bias) (+relu). f32 in/out, f16 math.
// 256 threads = 8 waves; block tile 128x64; wave tile 32x32 (2x2 wmma frags).
// LDS tiles are stored in fragment order so each fragment is 2x ds_load_b128.
// ---------------------------------------------------------------------------
__global__ __launch_bounds__(256)
void gemm_wmma_kernel(const float* __restrict__ A, int lda,
                      const float* __restrict__ Bm, int ldb,
                      const float* __restrict__ bias,
                      float* __restrict__ C, int ldc,
                      int M, int N, int K, int act)
{
    __shared__ _Float16 As[BM * LDS_STRIDE];   // row-major rows, natural K
    __shared__ _Float16 Bs[BN * LDS_STRIDE];   // column-major cols, natural K

    const int bm0  = blockIdx.x * BM;
    const int bn0  = blockIdx.y * BN;
    const int tid  = threadIdx.x;
    const int lane = tid & 31;
    const int wid  = tid >> 5;
    const int waveM = wid & 3;    // 0..3 -> 32-row strips
    const int waveN = wid >> 2;   // 0..1 -> 32-col strips
    const int l16  = lane & 15;
    const int hi   = lane >> 4;   // 0 or 1

    const bool fullM = (bm0 + BM) <= M;
    const bool fullN = (bn0 + BN) <= N;

    v8f acc[2][2] = {};

    for (int k0 = 0; k0 < K; k0 += BK) {
        const bool fullK = (k0 + BK) <= K;

        // ---- stage A tile: 128 rows x 32 K-halves, fragment-ordered ----
        if (fullM & fullK) {
            #pragma unroll
            for (int i = 0; i < 4; ++i) {
                int idx = i * 256 + tid;                 // 1024 quads
                int r = idx >> 3, c4 = (idx & 7) << 2;
                const float* ap = &A[(long)(bm0 + r) * lda + k0 + c4];
                float x0 = ap[0], x1 = ap[1], x2 = ap[2], x3 = ap[3];
                v4h hv = { (_Float16)x0, (_Float16)x1, (_Float16)x2, (_Float16)x3 };
                *(v4h*)&As[r * LDS_STRIDE + c4] = hv;   // ds_store_b64
            }
        } else {
            #pragma unroll
            for (int i = 0; i < 4; ++i) {
                int idx = i * 256 + tid;
                int r = idx >> 3, c4 = (idx & 7) << 2;
                int gr = bm0 + r;
                #pragma unroll
                for (int j = 0; j < 4; ++j) {
                    int gc = k0 + c4 + j;
                    float v = (gr < M && gc < K) ? A[(long)gr * lda + gc] : 0.0f;
                    As[r * LDS_STRIDE + c4 + j] = (_Float16)v;
                }
            }
        }

        // ---- stage B tile: 32 K x 64 N, stored column-major ----
        if (fullN & fullK) {
            #pragma unroll
            for (int i = 0; i < 2; ++i) {
                int idx = i * 256 + tid;                 // 512 quads
                int r = idx >> 4, c4 = (idx & 15) << 2;  // r = K row, c = N col
                const float* bp = &Bm[(long)(k0 + r) * ldb + bn0 + c4];
                float x0 = bp[0], x1 = bp[1], x2 = bp[2], x3 = bp[3];
                Bs[(c4 + 0) * LDS_STRIDE + r] = (_Float16)x0;
                Bs[(c4 + 1) * LDS_STRIDE + r] = (_Float16)x1;
                Bs[(c4 + 2) * LDS_STRIDE + r] = (_Float16)x2;
                Bs[(c4 + 3) * LDS_STRIDE + r] = (_Float16)x3;
            }
        } else {
            #pragma unroll
            for (int i = 0; i < 2; ++i) {
                int idx = i * 256 + tid;
                int r = idx >> 4, c4 = (idx & 15) << 2;
                int gk = k0 + r;
                #pragma unroll
                for (int j = 0; j < 4; ++j) {
                    int gc = bn0 + c4 + j;
                    float v = (gk < K && gc < N) ? Bm[(long)gk * ldb + gc] : 0.0f;
                    Bs[(c4 + j) * LDS_STRIDE + r] = (_Float16)v;
                }
            }
        }

        // prefetch next A tile (gfx1250 global_prefetch_b8)
        if (k0 + BK < K) {
            int pr = bm0 + (tid >> 1);
            if (pr < M) __builtin_prefetch(&A[(long)pr * lda + k0 + BK], 0, 1);
        }
        __syncthreads();

        // ---- fragments: contiguous 16 halves per lane -> 2x ds_load_b128 ----
        union Frag { v16h v; v8h h[2]; };
        Frag af[2], bf[2];
        #pragma unroll
        for (int mi = 0; mi < 2; ++mi) {
            const _Float16* src = &As[(waveM * 32 + mi * 16 + l16) * LDS_STRIDE + hi * 16];
            af[mi].h[0] = *(const v8h*)(src);
            af[mi].h[1] = *(const v8h*)(src + 8);
        }
        #pragma unroll
        for (int ni = 0; ni < 2; ++ni) {
            const _Float16* src = &Bs[(waveN * 32 + ni * 16 + l16) * LDS_STRIDE + hi * 16];
            bf[ni].h[0] = *(const v8h*)(src);
            bf[ni].h[1] = *(const v8h*)(src + 8);
        }
        #pragma unroll
        for (int mi = 0; mi < 2; ++mi)
            #pragma unroll
            for (int ni = 0; ni < 2; ++ni)
                acc[mi][ni] = __builtin_amdgcn_wmma_f32_16x16x32_f16(
                    false, af[mi].v, false, bf[ni].v,
                    (short)0, acc[mi][ni], false, false);
        __syncthreads();
    }

    // ---- epilogue: C/D layout -> VGPR i holds M = i (+8 for hi lanes), N = l16 ----
    if (fullM & fullN) {
        #pragma unroll
        for (int mi = 0; mi < 2; ++mi) {
            #pragma unroll
            for (int ni = 0; ni < 2; ++ni) {
                int gc = bn0 + waveN * 32 + ni * 16 + l16;
                float bv = bias ? bias[gc] : 0.0f;
                #pragma unroll
                for (int i = 0; i < 8; ++i) {
                    int gr = bm0 + waveM * 32 + mi * 16 + i + hi * 8;
                    float v = acc[mi][ni][i] + bv;
                    if (act == 1) v = fmaxf(v, 0.0f);
                    C[(long)gr * ldc + gc] = v;
                }
            }
        }
    } else {
        #pragma unroll
        for (int mi = 0; mi < 2; ++mi) {
            #pragma unroll
            for (int ni = 0; ni < 2; ++ni) {
                int gc = bn0 + waveN * 32 + ni * 16 + l16;
                float bv = (bias && gc < N) ? bias[gc] : 0.0f;
                #pragma unroll
                for (int i = 0; i < 8; ++i) {
                    int gr = bm0 + waveM * 32 + mi * 16 + i + hi * 8;
                    if (gr < M && gc < N) {
                        float v = acc[mi][ni][i] + bv;
                        if (act == 1) v = fmaxf(v, 0.0f);
                        C[(long)gr * ldc + gc] = v;
                    }
                }
            }
        }
    }
}

// ---------------------------------------------------------------------------
// Graph / elementwise helper kernels
// ---------------------------------------------------------------------------
__device__ __forceinline__ unsigned f2ord(float f) {
    unsigned u = __float_as_uint(f);
    return (u & 0x80000000u) ? ~u : (u | 0x80000000u);
}
__device__ __forceinline__ float ord2f(unsigned u) {
    unsigned v = (u & 0x80000000u) ? (u ^ 0x80000000u) : ~u;
    return __uint_as_float(v);
}

__global__ void fill_u32_kernel(unsigned* p, unsigned v, long n) {
    long i = blockIdx.x * (long)blockDim.x + threadIdx.x;
    if (i < n) p[i] = v;
}

__global__ void gat_logits_kernel(const float* __restrict__ h,
                                  const float* __restrict__ asrc,
                                  const float* __restrict__ adst,
                                  float* __restrict__ al, float* __restrict__ ar,
                                  int n, int C) {
    int i = blockIdx.x * blockDim.x + threadIdx.x;
    if (i >= n * HEADS) return;
    int node = i / HEADS, hd = i % HEADS;
    const float* hp = h + (long)node * HEADS * C + hd * C;
    const float* as_ = asrc + hd * C;
    const float* ad_ = adst + hd * C;
    float s1 = 0.f, s2 = 0.f;
    for (int c = 0; c < C; ++c) { float v = hp[c]; s1 += v * as_[c]; s2 += v * ad_[c]; }
    al[i] = s1; ar[i] = s2;
}

__device__ __forceinline__ void edge_sd(const int* ei, long E, long idx, int& s, int& d) {
    if (idx < E) { s = ei[idx]; d = ei[E + idx]; }
    else         { s = d = (int)(idx - E); }
}

__global__ void gat_edge_max_kernel(const int* __restrict__ ei, int E, int n,
                                    const float* __restrict__ al,
                                    const float* __restrict__ ar,
                                    unsigned* __restrict__ mmax) {
    long i = blockIdx.x * (long)blockDim.x + threadIdx.x;
    long tot = (long)E + n;
    if (i >= tot) return;
    int s, d; edge_sd(ei, E, i, s, d);
    #pragma unroll
    for (int h = 0; h < HEADS; ++h) {
        float e = al[s * HEADS + h] + ar[d * HEADS + h];
        e = (e > 0.f) ? e : 0.2f * e;
        atomicMax(&mmax[d * HEADS + h], f2ord(e));
    }
}

__global__ void gat_edge_exp_kernel(const int* __restrict__ ei, int E, int n,
                                    const float* __restrict__ al,
                                    const float* __restrict__ ar,
                                    const unsigned* __restrict__ mmax,
                                    float* __restrict__ ex, float* __restrict__ den) {
    long i = blockIdx.x * (long)blockDim.x + threadIdx.x;
    long tot = (long)E + n;
    if (i >= tot) return;
    int s, d; edge_sd(ei, E, i, s, d);
    #pragma unroll
    for (int h = 0; h < HEADS; ++h) {
        float e = al[s * HEADS + h] + ar[d * HEADS + h];
        e = (e > 0.f) ? e : 0.2f * e;
        float x = __expf(e - ord2f(mmax[d * HEADS + h]));
        ex[i * HEADS + h] = x;
        atomicAdd(&den[d * HEADS + h], x);
    }
}

__global__ void gat_scatter_kernel(const int* __restrict__ ei, int E, int n,
                                   const float* __restrict__ hmat,
                                   const float* __restrict__ ex,
                                   const float* __restrict__ den,
                                   float* __restrict__ out, int C) {
    long tot = ((long)E + n) * HEADS;
    long i = blockIdx.x * (long)blockDim.x + threadIdx.x;
    if (i >= tot) return;
    long eidx = i / HEADS; int h = (int)(i % HEADS);
    int s, d; edge_sd(ei, E, eidx, s, d);
    float alpha = ex[eidx * HEADS + h] / den[d * HEADS + h];
    const float* hp = hmat + (long)s * HEADS * C + h * C;
    float*       op = out  + (long)d * HEADS * C + h * C;
    for (int c = 0; c < C; ++c) atomicAdd(&op[c], alpha * hp[c]);
}

__global__ void bias_act_kernel(float* __restrict__ x, const float* __restrict__ bias,
                                long rows, int cols, int act) {
    long i = blockIdx.x * (long)blockDim.x + threadIdx.x;
    if (i >= rows * cols) return;
    int c = (int)(i % cols);
    float v = x[i] + bias[c];
    if (act) v = fmaxf(v, 0.0f);
    x[i] = v;
}

__global__ void deg_kernel(const int* __restrict__ ei, int E, int n, float* __restrict__ deg) {
    long i = blockIdx.x * (long)blockDim.x + threadIdx.x;
    long tot = (long)E + n;
    if (i >= tot) return;
    int s, d; edge_sd(ei, E, i, s, d);
    (void)s;
    atomicAdd(&deg[d], 1.0f);
}

__global__ void dinv_kernel(float* __restrict__ deg, int n) {
    int i = blockIdx.x * blockDim.x + threadIdx.x;
    if (i >= n) return;
    float v = deg[i];
    deg[i] = (v > 0.f) ? rsqrtf(v) : 0.f;
}

__global__ void gcn_scatter_kernel(const int* __restrict__ ei, int E, int n,
                                   const float* __restrict__ dinv,
                                   const float* __restrict__ hmat,
                                   float* __restrict__ out, int C) {
    long tot = (long)E + n;
    long i = blockIdx.x * (long)blockDim.x + threadIdx.x;
    if (i >= tot) return;
    int s, d; edge_sd(ei, E, i, s, d);
    float norm = dinv[s] * dinv[d];
    const float* hp = hmat + (long)s * C;
    float*       op = out  + (long)d * C;
    for (int c = 0; c < C; ++c) atomicAdd(&op[c], norm * hp[c]);
}

__global__ void count_kernel(const int* __restrict__ batch, int n, float* __restrict__ cnt) {
    int i = blockIdx.x * blockDim.x + threadIdx.x;
    if (i < n) atomicAdd(&cnt[batch[i]], 1.0f);
}

__global__ void pool_sum_kernel(const float* __restrict__ h, const int* __restrict__ batch,
                                int rows, int row0, int cols,
                                float* __restrict__ xc, int xcld, int coff) {
    long i = blockIdx.x * (long)blockDim.x + threadIdx.x;
    if (i >= (long)rows * cols) return;
    int r = (int)(i / cols), c = (int)(i % cols);
    int b = batch[row0 + r];
    atomicAdd(&xc[(long)b * xcld + coff + c], h[(long)r * cols + c]);
}

__global__ void pool_div_kernel(float* __restrict__ xc, const float* __restrict__ cntM,
                                const float* __restrict__ cntP) {
    int i = blockIdx.x * blockDim.x + threadIdx.x;
    if (i >= NB * 8 * OD) return;
    int b = i / (8 * OD), c = i % (8 * OD);
    float den = (c < 4 * OD) ? cntM[b] : cntP[b];
    xc[i] = xc[i] / fmaxf(den, 1.0f);
}

__global__ void softplus_kernel(float* __restrict__ x, long n) {
    long i = blockIdx.x * (long)blockDim.x + threadIdx.x;
    if (i >= n) return;
    float v = x[i] - 5.0f;
    x[i] = (v > 20.f) ? v : log1pf(__expf(v));
}

__device__ __forceinline__ unsigned pcg_hash(unsigned v) {
    unsigned s = v * 747796405u + 2891336453u;
    unsigned w = ((s >> ((s >> 28) + 4u)) ^ s) * 277803737u;
    return (w >> 22) ^ w;
}

__global__ void latent_kernel(const float* __restrict__ mean, const float* __restrict__ stdv,
                              float* __restrict__ lat, long n) {
    long i = blockIdx.x * (long)blockDim.x + threadIdx.x;
    if (i >= n) return;
    unsigned w1 = pcg_hash((unsigned)i * 2654435769u + 42u);
    unsigned w2 = pcg_hash((unsigned)i ^ 0x9E3779B9u);
    float u1 = (w1 + 1u) * 2.3283064365386963e-10f;
    float u2 = w2 * 2.3283064365386963e-10f;
    float eps = sqrtf(-2.f * logf(u1)) * __cosf(6.2831853f * u2);
    lat[i] = mean[i] + stdv[i] * eps;
}

__global__ void final_out_kernel(const float* __restrict__ h3, const float* __restrict__ W,
                                 const float* __restrict__ b, float* __restrict__ out,
                                 int rows, int K) {
    int r = blockIdx.x * blockDim.x + threadIdx.x;
    if (r >= rows) return;
    float s = b[0];
    for (int k = 0; k < K; ++k) s += h3[(long)r * K + k] * W[k];
    out[r] = s;
}

// ---------------------------------------------------------------------------
// Host side
// ---------------------------------------------------------------------------
static inline void gemm(hipStream_t s, const float* A, int lda, const float* B, int ldb,
                        const float* bias, float* C, int ldc, int M, int N, int K, int act) {
    dim3 g((M + BM - 1) / BM, (N + BN - 1) / BN), b(256);
    gemm_wmma_kernel<<<g, b, 0, s>>>(A, lda, B, ldb, bias, C, ldc, M, N, K, act);
}

static inline int blks(long n, int t) { return (int)((n + t - 1) / t); }

struct Ws {
    float *attH, *attOut, *al, *ar, *den, *ex, *dinv, *hc, *x3, *hcat, *hf1, *hf2;
    unsigned* mmax;
    float *xc, *cntM, *cntP, *h1, *h3;
};

static void run_branch(hipStream_t st, Ws& w,
                       const float* x, const int* ei, const int* batch,
                       int n, int E, int F,
                       const float* gW, const float* as_, const float* ad_, const float* gb,
                       const float* cW, const float* cb,
                       const float* W1, const float* W3,
                       const float* f1W, const float* f1b,
                       const float* f2W, const float* f2b,
                       float* cnt, int coff)
{
    const int F3 = 3 * F;
    const int HW = F3 + OD;           // hcat width
    const long EN = (long)E + n;

    // GAT lin: attH = x @ gW
    gemm(st, x, F, gW, F3, nullptr, w.attH, F3, n, F3, F, 0);
    // per-node per-head logits
    gat_logits_kernel<<<blks((long)n * HEADS, 256), 256, 0, st>>>(w.attH, as_, ad_, w.al, w.ar, n, F);
    // segment max / exp / denom
    fill_u32_kernel<<<blks((long)n * HEADS, 256), 256, 0, st>>>(w.mmax, 0x007FFFFFu, (long)n * HEADS); // enc(-inf)
    hipMemsetAsync(w.den, 0, (size_t)n * HEADS * sizeof(float), st);
    gat_edge_max_kernel<<<blks(EN, 256), 256, 0, st>>>(ei, E, n, w.al, w.ar, w.mmax);
    gat_edge_exp_kernel<<<blks(EN, 256), 256, 0, st>>>(ei, E, n, w.al, w.ar, w.mmax, w.ex, w.den);
    // attention-weighted scatter
    hipMemsetAsync(w.attOut, 0, (size_t)n * F3 * sizeof(float), st);
    gat_scatter_kernel<<<blks(EN * HEADS, 256), 256, 0, st>>>(ei, E, n, w.attH, w.ex, w.den, w.attOut, F);
    bias_act_kernel<<<blks((long)n * F3, 256), 256, 0, st>>>(w.attOut, gb, n, F3, 1);  // x1

    // GCN
    hipMemsetAsync(w.dinv, 0, (size_t)n * sizeof(float), st);
    deg_kernel<<<blks(EN, 256), 256, 0, st>>>(ei, E, n, w.dinv);
    dinv_kernel<<<blks(n, 256), 256, 0, st>>>(w.dinv, n);
    gemm(st, w.attOut, F3, cW, OD, nullptr, w.hc, OD, n, OD, F3, 0);
    hipMemsetAsync(w.x3, 0, (size_t)n * OD * sizeof(float), st);
    gcn_scatter_kernel<<<blks(EN, 256), 256, 0, st>>>(ei, E, n, w.dinv, w.hc, w.x3, OD);
    bias_act_kernel<<<blks((long)n * OD, 256), 256, 0, st>>>(w.x3, cb, n, OD, 1);      // x3

    // hcat = [x1 @ W1 | x3 @ W3]
    gemm(st, w.attOut, F3, W1, F3, nullptr, w.hcat, HW, n, F3, F3, 0);
    gemm(st, w.x3, OD, W3, OD, nullptr, w.hcat + F3, HW, n, OD, OD, 0);

    // counts for mean-pool
    count_kernel<<<blks(n, 256), 256, 0, st>>>(batch, n, cnt);

    // chunked fc1 -> fc2 -> pool
    for (int r0 = 0; r0 < n; r0 += CHUNK) {
        int rows = (n - r0 < CHUNK) ? (n - r0) : CHUNK;
        gemm(st, w.hcat + (long)r0 * HW, HW, f1W, 2048, f1b, w.hf1, 2048, rows, 2048, HW, 1);
        gemm(st, w.hf1, 2048, f2W, 512, f2b, w.hf2, 512, rows, 512, 2048, 1);
        pool_sum_kernel<<<blks((long)rows * 512, 256), 256, 0, st>>>(
            w.hf2, batch, rows, r0, 512, w.xc, 8 * OD, coff);
    }
}

extern "C" void kernel_launch(void* const* d_in, const int* in_sizes, int n_in,
                              void* d_out, int out_size, void* d_ws, size_t ws_size,
                              hipStream_t stream)
{
    (void)in_sizes; (void)n_in; (void)out_size; (void)ws_size;

    const float* mol_x    = (const float*)d_in[0];
    const int*   mol_ei   = (const int*)d_in[1];
    const int*   mol_b    = (const int*)d_in[2];
    const float* pro_x    = (const float*)d_in[3];
    const int*   pro_ei   = (const int*)d_in[4];
    const int*   pro_b    = (const int*)d_in[5];
    const float* mgW = (const float*)d_in[6],  *mas = (const float*)d_in[7],
               *mad = (const float*)d_in[8],  *mgb = (const float*)d_in[9],
               *mcW = (const float*)d_in[10], *mcb = (const float*)d_in[11],
               *mW1 = (const float*)d_in[12], *mW3 = (const float*)d_in[13],
               *mf1W = (const float*)d_in[14], *mf1b = (const float*)d_in[15],
               *mf2W = (const float*)d_in[16], *mf2b = (const float*)d_in[17];
    const float* pgW = (const float*)d_in[18], *pas = (const float*)d_in[19],
               *pad = (const float*)d_in[20], *pgb = (const float*)d_in[21],
               *pcW = (const float*)d_in[22], *pcb = (const float*)d_in[23],
               *pW1 = (const float*)d_in[24], *pW3 = (const float*)d_in[25],
               *pf1W = (const float*)d_in[26], *pf1b = (const float*)d_in[27],
               *pf2W = (const float*)d_in[28], *pf2b = (const float*)d_in[29];
    const float* emW = (const float*)d_in[30], *emb = (const float*)d_in[31],
               *esW = (const float*)d_in[32], *esb = (const float*)d_in[33],
               *fc1W = (const float*)d_in[34], *fc1b = (const float*)d_in[35],
               *fc3W = (const float*)d_in[36], *fc3b = (const float*)d_in[37],
               *outW = (const float*)d_in[38], *outb = (const float*)d_in[39];

    // ---- workspace arena (sized for the larger of the two branches) ----
    char* p = (char*)d_ws;
    auto bump = [&](size_t bytes) -> char* {
        char* r = p; p += (bytes + 255) & ~(size_t)255; return r;
    };
    const long maxN = N_PRO;                                   // 120000
    const long maxAttH = (long)N_PRO * 3 * FP > (long)N_MOL * 3 * FM
                       ? (long)N_PRO * 3 * FP : (long)N_MOL * 3 * FM;   // 19.44M
    const long maxHcat = (long)N_PRO * (3 * FP + OD);                   // 34.8M
    const long maxEN   = (long)E_PRO + N_PRO;

    Ws w;
    w.attH   = (float*)bump(maxAttH * 4);
    w.attOut = (float*)bump(maxAttH * 4);
    w.al     = (float*)bump(maxN * HEADS * 4);
    w.ar     = (float*)bump(maxN * HEADS * 4);
    w.mmax   = (unsigned*)bump(maxN * HEADS * 4);
    w.den    = (float*)bump(maxN * HEADS * 4);
    w.ex     = (float*)bump(maxEN * HEADS * 4);
    w.dinv   = (float*)bump(maxN * 4);
    w.hc     = (float*)bump(maxN * OD * 4);
    w.x3     = (float*)bump(maxN * OD * 4);
    w.hcat   = (float*)bump(maxHcat * 4);
    w.hf1    = (float*)bump((long)CHUNK * 2048 * 4);
    w.hf2    = (float*)bump((long)CHUNK * 512 * 4);
    w.xc     = (float*)bump((long)NB * 8 * OD * 4);
    w.cntM   = (float*)bump(NB * 4);
    w.cntP   = (float*)bump(NB * 4);
    w.h1     = (float*)bump((long)NB * 2048 * 4);
    w.h3     = (float*)bump((long)NB * 512 * 4);

    hipMemsetAsync(w.xc, 0, (size_t)NB * 8 * OD * 4, stream);
    hipMemsetAsync(w.cntM, 0, NB * 4, stream);
    hipMemsetAsync(w.cntP, 0, NB * 4, stream);

    // ---- branches ----
    run_branch(stream, w, mol_x, mol_ei, mol_b, N_MOL, E_MOL, FM,
               mgW, mas, mad, mgb, mcW, mcb, mW1, mW3, mf1W, mf1b, mf2W, mf2b,
               w.cntM, 0);
    run_branch(stream, w, pro_x, pro_ei, pro_b, N_PRO, E_PRO, FP,
               pgW, pas, pad, pgb, pcW, pcb, pW1, pW3, pf1W, pf1b, pf2W, pf2b,
               w.cntP, 4 * OD);
    pool_div_kernel<<<blks((long)NB * 8 * OD, 256), 256, 0, stream>>>(w.xc, w.cntM, w.cntP);

    // ---- head: write tuple (out, enc_mean, enc_std, latent) into d_out ----
    float* o_out  = (float*)d_out;                       // [256]
    float* o_mean = o_out + NB;                          // [256,1024]
    float* o_std  = o_mean + (long)NB * 8 * OD;          // [256,1024]
    float* o_lat  = o_std  + (long)NB * 8 * OD;          // [256,1024]
    const int HD = 8 * OD;                               // 1024
    const long HN = (long)NB * HD;

    gemm(stream, w.xc, HD, emW, HD, emb, o_mean, HD, NB, HD, HD, 0);
    gemm(stream, w.xc, HD, esW, HD, esb, o_std,  HD, NB, HD, HD, 0);
    softplus_kernel<<<blks(HN, 256), 256, 0, stream>>>(o_std, HN);
    latent_kernel<<<blks(HN, 256), 256, 0, stream>>>(o_mean, o_std, o_lat, HN);

    gemm(stream, o_lat, HD, fc1W, 2048, fc1b, w.h1, 2048, NB, 2048, HD, 1);
    gemm(stream, w.h1, 2048, fc3W, 512, fc3b, w.h3, 512, NB, 512, 2048, 1);
    final_out_kernel<<<1, 256, 0, stream>>>(w.h3, outW, outb, o_out, NB, 512);
}
